// QuantizedDiffusionPredictor_52828097741018
// MI455X (gfx1250) — matrix operand order repaired and checked by reading, not compiled, not run on hardware
//
#include <hip/hip_runtime.h>
#include <hip/hip_bf16.h>
#include <math.h>

#define DEV static __device__ __forceinline__

typedef __bf16 bf16;
typedef __attribute__((ext_vector_type(16))) __bf16 v16bf;
typedef __attribute__((ext_vector_type(8)))  __bf16 v8bf;
typedef __attribute__((ext_vector_type(8)))  float  v8f;

union FragU { v16bf v; v8bf h[2]; };

DEV v8bf cvt8(const float* p) {
  float4 a = *(const float4*)p;
  float4 b = *(const float4*)(p + 4);
  v8bf r;
  r[0] = (bf16)a.x; r[1] = (bf16)a.y; r[2] = (bf16)a.z; r[3] = (bf16)a.w;
  r[4] = (bf16)b.x; r[5] = (bf16)b.y; r[6] = (bf16)b.z; r[7] = (bf16)b.w;
  return r;
}

DEV v8f vzero() { v8f z; __builtin_memset(&z, 0, sizeof(z)); return z; }

DEV v8f wmma_bf16(v16bf a, v16bf b, v8f c) {
  return __builtin_amdgcn_wmma_f32_16x16x32_bf16(false, a, false, b, (short)0, c, false, false);
}

// async global(bf16) -> LDS, 16 bytes per active lane (ASYNCcnt tracked)
DEV void async_b128(const bf16* g, void* lds) {
  unsigned l = (unsigned)(size_t)lds;     // low 32 bits of flat LDS ptr = LDS offset
  asm volatile("global_load_async_to_lds_b128 %0, %1, off"
               :: "v"(l), "v"(g) : "memory");
}
DEV void wait_async0() { asm volatile("s_wait_asynccnt 0x0" ::: "memory"); }

DEV float block_sum(float v, volatile float* red8, int tid) {
  #pragma unroll
  for (int off = 16; off >= 1; off >>= 1) v += __shfl_xor(v, off, 32);
  __syncthreads();
  if ((tid & 31) == 0) red8[tid >> 5] = v;
  __syncthreads();
  float s = 0.f;
  #pragma unroll
  for (int i = 0; i < 8; ++i) s += red8[i];
  return s;
}

// ---------------------------------------------------------------------------
// bf16-WMMA GEMM: Out[M,N] = act(A[M,Kd] @ W[N,Kd]^T + bias) (+Res)
// Double-buffered LDS; W (bf16) and bf16-A staged via async DMA issued one
// K-chunk ahead of the WMMA stream. ACT: 0=none, 1=gelu(exact).
// 256 thr / 8 waves, tile 128x128, KB=64.
// ---------------------------------------------------------------------------
template<int ACT, bool HAS_RES, bool ABF, bool OUT_BF>
__global__ __launch_bounds__(256)
void gemm_bf16(const void* __restrict__ Ap, const bf16* __restrict__ W,
               const float* __restrict__ bias, const float* __restrict__ Res,
               void* __restrict__ Outp, int M, int N, int Kd) {
  __shared__ bf16 As[2][128][72];
  __shared__ bf16 Ws[2][128][72];
  const float* Af = (const float*)Ap;
  const bf16*  Ab = (const bf16*)Ap;
  float* Outf = (float*)Outp;
  bf16*  Outb = (bf16*)Outp;
  const int tid = threadIdx.x;
  const int lane = tid & 31, wave = tid >> 5;
  const int waveM = wave >> 2, waveN = wave & 3;   // 2 x 4 wave grid
  const int lhi = lane >> 4, llo = lane & 15;
  const int bm = blockIdx.y * 128, bn = blockIdx.x * 128;

  auto stage = [&](int buf, int kb) {
    #pragma unroll
    for (int i = 0; i < 4; ++i) {        // W tile: 1024 x 16B chunks
      int idx8 = tid + i * 256;
      int row = idx8 >> 3, c8 = (idx8 & 7) << 3;
      async_b128(W + (size_t)(bn + row) * Kd + kb + c8, &Ws[buf][row][c8]);
    }
    if (ABF) {
      #pragma unroll
      for (int i = 0; i < 4; ++i) {
        int idx8 = tid + i * 256;
        int row = idx8 >> 3, c8 = (idx8 & 7) << 3;
        async_b128(Ab + (size_t)(bm + row) * Kd + kb + c8, &As[buf][row][c8]);
      }
    } else {
      #pragma unroll
      for (int i = 0; i < 8; ++i) {      // f32 A: load + convert on stage
        int idx4 = tid + i * 256;
        int row = idx4 >> 4, c4 = (idx4 & 15) * 4;
        const float* pa = Af + (size_t)(bm + row) * Kd + kb + c4;
        float4 va = *(const float4*)pa;
        As[buf][row][c4+0] = (bf16)va.x; As[buf][row][c4+1] = (bf16)va.y;
        As[buf][row][c4+2] = (bf16)va.z; As[buf][row][c4+3] = (bf16)va.w;
      }
    }
  };

  v8f acc[4][2];
  #pragma unroll
  for (int i = 0; i < 4; ++i)
    #pragma unroll
    for (int j = 0; j < 2; ++j) acc[i][j] = vzero();

  stage(0, 0);
  wait_async0();
  __syncthreads();

  const int nk = Kd >> 6;
  for (int k = 0; k < nk; ++k) {
    const int cur = k & 1;
    if (k + 1 < nk) stage(cur ^ 1, (k + 1) * 64);   // prefetch next chunk (DMA)
    #pragma unroll
    for (int ks = 0; ks < 2; ++ks) {
      FragU af[4], bw[2];
      #pragma unroll
      for (int mf = 0; mf < 4; ++mf) {
        int r  = waveM * 64 + mf * 16 + llo;
        int ko = ks * 32 + 8 * lhi;
        af[mf].h[0] = *(const v8bf*)&As[cur][r][ko];
        af[mf].h[1] = *(const v8bf*)&As[cur][r][ko + 16];
      }
      #pragma unroll
      for (int nf = 0; nf < 2; ++nf) {
        int r  = waveN * 32 + nf * 16 + llo;
        int ko = ks * 32 + 8 * lhi;
        bw[nf].h[0] = *(const v8bf*)&Ws[cur][r][ko];
        bw[nf].h[1] = *(const v8bf*)&Ws[cur][r][ko + 16];
      }
      #pragma unroll
      for (int mf = 0; mf < 4; ++mf)
        #pragma unroll
        for (int nf = 0; nf < 2; ++nf)
          acc[mf][nf] = wmma_bf16(af[mf].v, bw[nf].v, acc[mf][nf]);
    }
    wait_async0();        // next-chunk DMA landed during the WMMA burst
    __syncthreads();
  }

  #pragma unroll
  for (int mf = 0; mf < 4; ++mf)
    #pragma unroll
    for (int nf = 0; nf < 2; ++nf)
      #pragma unroll
      for (int r = 0; r < 8; ++r) {
        int m = bm + waveM * 64 + mf * 16 + r + 8 * lhi;
        int n = bn + waveN * 32 + nf * 16 + llo;
        float v = acc[mf][nf][r] + (bias ? bias[n] : 0.f);
        if (ACT == 1) v = 0.5f * v * (1.0f + erff(v * 0.70710678118f));
        if (HAS_RES) v += Res[(size_t)m * N + n];
        if (OUT_BF) Outb[(size_t)m * N + n] = (bf16)v;
        else        Outf[(size_t)m * N + n] = v;
      }
}

// ---------------------------------------------------------------------------
// Flash attention: one block per (seq, head). qkv is bf16 -> Q/K frags are
// raw 16B global loads; V transposed in LDS; online softmax (shfl row
// reductions); P through per-wave LDS tile. Writes bf16 (feeds o-proj GEMM).
// ---------------------------------------------------------------------------
__global__ __launch_bounds__(256)
void attn_kernel(const bf16* __restrict__ qkv, bf16* __restrict__ ao) {
  __shared__ bf16 Vt[64][264];       // [dh][token]
  __shared__ bf16 Pw[8][16][40];     // per-wave 16x32 P chunk
  const int bfi = blockIdx.x, head = blockIdx.y;
  const int tbase = bfi * 256;
  const int tid = threadIdx.x, lane = tid & 31, wave = tid >> 5;
  const int lhi = lane >> 4, llo = lane & 15;
  const int qoff = head * 64;

  for (int i = 0; i < 64; ++i) {     // stage V transposed
    int idx = i * 256 + tid;
    int t = idx >> 6, k = idx & 63;
    Vt[k][t] = qkv[(size_t)(tbase + t) * 3072 + 2048 + qoff + k];
  }
  __syncthreads();

  for (int qt2 = 0; qt2 < 2; ++qt2) {
    int qt = wave * 2 + qt2;
    FragU aq[2];
    int qrow = tbase + qt * 16 + llo;
    #pragma unroll
    for (int ks = 0; ks < 2; ++ks) {
      const bf16* p0 = qkv + (size_t)qrow * 3072 + qoff + ks * 32 + 8 * lhi;
      aq[ks].h[0] = *(const v8bf*)p0;
      aq[ks].h[1] = *(const v8bf*)(p0 + 16);
    }
    float rmax[8], rsum[8];
    v8f oacc[4];
    #pragma unroll
    for (int r = 0; r < 8; ++r) { rmax[r] = -3.0e38f; rsum[r] = 0.f; }
    #pragma unroll
    for (int nt = 0; nt < 4; ++nt) oacc[nt] = vzero();

    for (int kc = 0; kc < 8; ++kc) {       // 32 keys per chunk
      v8f s[2];
      #pragma unroll
      for (int h2 = 0; h2 < 2; ++h2) {
        int kt = kc * 2 + h2;
        v8f sa = vzero();
        #pragma unroll
        for (int ks = 0; ks < 2; ++ks) {
          FragU bk;
          const bf16* pk = qkv + (size_t)(tbase + kt * 16 + llo) * 3072
                           + 1024 + qoff + ks * 32 + 8 * lhi;
          bk.h[0] = *(const v8bf*)pk;
          bk.h[1] = *(const v8bf*)(pk + 16);
          sa = wmma_bf16(aq[ks].v, bk.v, sa);
        }
        s[h2] = sa;
      }
      #pragma unroll
      for (int r = 0; r < 8; ++r) {
        float cm = fmaxf(s[0][r], s[1][r]);
        #pragma unroll
        for (int off = 8; off >= 1; off >>= 1)
          cm = fmaxf(cm, __shfl_xor(cm, off, 32));
        float nm = fmaxf(rmax[r], cm);
        float corr = __expf((rmax[r] - nm) * 0.125f);
        rmax[r] = nm;
        int prow = r + 8 * lhi;
        float ps = 0.f;
        #pragma unroll
        for (int h2 = 0; h2 < 2; ++h2) {
          float p = __expf((s[h2][r] - nm) * 0.125f);
          Pw[wave][prow][h2 * 16 + llo] = (bf16)p;
          ps += p;
        }
        #pragma unroll
        for (int off = 8; off >= 1; off >>= 1)
          ps += __shfl_xor(ps, off, 32);
        rsum[r] = rsum[r] * corr + ps;
        #pragma unroll
        for (int nt = 0; nt < 4; ++nt) oacc[nt][r] *= corr;
      }
      __syncthreads();
      FragU ap;
      ap.h[0] = *(const v8bf*)&Pw[wave][llo][8 * lhi];
      ap.h[1] = *(const v8bf*)&Pw[wave][llo][16 + 8 * lhi];
      #pragma unroll
      for (int nt = 0; nt < 4; ++nt) {
        FragU bv;
        bv.h[0] = *(const v8bf*)&Vt[nt * 16 + llo][kc * 32 + 8 * lhi];
        bv.h[1] = *(const v8bf*)&Vt[nt * 16 + llo][kc * 32 + 8 * lhi + 16];
        oacc[nt] = wmma_bf16(ap.v, bv.v, oacc[nt]);
      }
      __syncthreads();
    }
    #pragma unroll
    for (int nt = 0; nt < 4; ++nt)
      #pragma unroll
      for (int r = 0; r < 8; ++r) {
        int token = tbase + qt * 16 + r + 8 * lhi;
        int col = qoff + nt * 16 + llo;
        ao[(size_t)token * 1024 + col] = (bf16)(oacc[nt][r] / rsum[r]);
      }
  }
}

// ---------------------------------------------------------------------------
// LayerNorm over D=1024, one block per token; bf16 out (GEMM operand).
// ---------------------------------------------------------------------------
__global__ __launch_bounds__(256)
void ln_kernel(const float* __restrict__ in, const float* __restrict__ sc,
               const float* __restrict__ bi, bf16* __restrict__ out) {
  __shared__ float red8[8];
  const int t = blockIdx.x, tid = threadIdx.x;
  const float* row = in + (size_t)t * 1024;
  float x[4], s = 0.f, s2 = 0.f;
  #pragma unroll
  for (int i = 0; i < 4; ++i) {
    x[i] = row[tid + i * 256];
    s += x[i]; s2 += x[i] * x[i];
  }
  s  = block_sum(s,  red8, tid);
  s2 = block_sum(s2, red8, tid);
  float mean = s * (1.f / 1024.f);
  float var  = s2 * (1.f / 1024.f) - mean * mean;
  float rstd = rsqrtf(var + 1e-5f);
  #pragma unroll
  for (int i = 0; i < 4; ++i) {
    int d = tid + i * 256;
    out[(size_t)t * 1024 + d] = (bf16)((x[i] - mean) * rstd * sc[d] + bi[d]);
  }
}

// ---------------------------------------------------------------------------
// GroupNorm(32) + affine + SiLU; one block per (b, g); bf16 out.
// ---------------------------------------------------------------------------
__global__ __launch_bounds__(256)
void gn_silu_kernel(const float* __restrict__ hf, const float* __restrict__ gns,
                    const float* __restrict__ gnb, bf16* __restrict__ out) {
  __shared__ float red8[8];
  const int b = blockIdx.x >> 5, g = blockIdx.x & 31;
  const int tid = threadIdx.x;
  float s = 0.f, s2 = 0.f;
  for (int i = tid; i < 131072; i += 256) {
    int tt = i >> 5, dd = i & 31;
    float x = hf[(size_t)(b * 4096 + tt) * 1024 + g * 32 + dd];
    s += x; s2 += x * x;
  }
  s  = block_sum(s,  red8, tid);
  s2 = block_sum(s2, red8, tid);
  float mean = s * (1.f / 131072.f);
  float var  = s2 * (1.f / 131072.f) - mean * mean;
  float rstd = rsqrtf(var + 1e-5f);
  for (int i = tid; i < 131072; i += 256) {
    int tt = i >> 5, dd = i & 31;
    int d = g * 32 + dd;
    size_t o = (size_t)(b * 4096 + tt) * 1024 + d;
    float y = (hf[o] - mean) * rstd * gns[d] + gnb[d];
    out[o] = (bf16)(y / (1.f + __expf(-y)));
  }
}

// f32 -> bf16 bulk convert (weights / codebook, once per use)
__global__ void cvt_bf16_kernel(const float* __restrict__ s, bf16* __restrict__ d, int n) {
  int i = blockIdx.x * 256 + threadIdx.x;
  int stride = gridDim.x * 256;
  for (; i < n; i += stride) d[i] = (bf16)s[i];
}

// ---------------------------------------------------------------------------
// Timestep embedding chain (tiny)
// ---------------------------------------------------------------------------
__global__ void te_raw_kernel(const int* __restrict__ ts, float* __restrict__ teraw,
                              float* __restrict__ lacc) {
  int i = blockIdx.x * 256 + threadIdx.x;
  if (i == 0) lacc[0] = 0.f;
  if (i >= 1024) return;
  int b = i >> 9, j = i & 511;
  float freq = __expf(-logf(10000.f) * (float)j / 511.f);
  float a = (float)ts[b] * freq;
  teraw[b * 1024 + j] = __sinf(a);
  teraw[b * 1024 + 512 + j] = __cosf(a);
}

__global__ void ts_mlp1_kernel(const float* __restrict__ teraw, const float* __restrict__ tw1,
                               const float* __restrict__ tb1, float* __restrict__ te1) {
  int i = blockIdx.x * 256 + threadIdx.x;
  if (i >= 8192) return;
  int b = i >> 12, j = i & 4095;
  const float* xx = teraw + b * 1024;
  const float* ww = tw1 + (size_t)j * 1024;
  float s = tb1[j];
  for (int k = 0; k < 1024; ++k) s += xx[k] * ww[k];
  te1[i] = s / (1.f + __expf(-s));
}

__global__ void ts_mlp2_kernel(const float* __restrict__ te1, const float* __restrict__ tw2,
                               const float* __restrict__ tb2, float* __restrict__ te) {
  int i = blockIdx.x * 256 + threadIdx.x;
  if (i >= 2048) return;
  int b = i >> 10, j = i & 1023;
  const float* xx = te1 + b * 4096;
  const float* ww = tw2 + (size_t)j * 4096;
  float s = tb2[j];
  for (int k = 0; k < 4096; ++k) s += xx[k] * ww[k];
  te[i] = s;
}

// input proj (C_IN=16) + te + pos -> token tensor hf[8192][1024]
__global__ void tokens_kernel(const float* __restrict__ x, const float* __restrict__ in_w,
                              const float* __restrict__ in_b, const float* __restrict__ te,
                              const float* __restrict__ pos, float* __restrict__ hf) {
  int i = blockIdx.x * 256 + threadIdx.x;
  int t = i >> 10, d = i & 1023;
  int b = t >> 12, f = (t >> 8) & 15, hw = t & 255;
  float s = in_b[d];
  #pragma unroll
  for (int c = 0; c < 16; ++c)
    s += x[((b * 16 + c) * 16 + f) * 256 + hw] * in_w[d * 16 + c];
  hf[i] = s + te[b * 1024 + d] + pos[f * 1024 + d];
}

__global__ void cnorm_kernel(const float* __restrict__ cb, float* __restrict__ cn) {
  int i = blockIdx.x * 256 + threadIdx.x;
  if (i >= 8192) return;
  const float* r = cb + (size_t)i * 512;
  float s = 0.f;
  for (int k = 0; k < 512; ++k) s += r[k] * r[k];
  cn[i] = s;
}

// ---------------------------------------------------------------------------
// VQ argmin: dist = |c|^2 - 2 z.c. 64-code chunks of the bf16 codebook are
// async-DMA'd into LDS and shared by all 8 waves (8x L2 traffic cut).
// Each wave owns 16 tokens; WMMA over E=512 per 16-code tile.
// ---------------------------------------------------------------------------
__global__ __launch_bounds__(256)
void vq_argmin_kernel(const float* __restrict__ z, const bf16* __restrict__ cbb,
                      const float* __restrict__ cn, int* __restrict__ idx) {
  __shared__ bf16 Cs[64][520];       // 64 codes x 512 (pad 8 halves)
  const int tid = threadIdx.x, lane = tid & 31, wave = tid >> 5;
  const int lhi = lane >> 4, llo = lane & 15;
  const int tbase = blockIdx.x * 128 + wave * 16;
  FragU az[16];
  #pragma unroll
  for (int ks = 0; ks < 16; ++ks) {
    const float* p = z + (size_t)(tbase + llo) * 512 + ks * 32 + 8 * lhi;
    az[ks].h[0] = cvt8(p);
    az[ks].h[1] = cvt8(p + 16);
  }
  float mv[8]; int mi[8];
  #pragma unroll
  for (int r = 0; r < 8; ++r) { mv[r] = 3.4e38f; mi[r] = 0; }

  for (int nc = 0; nc < 128; ++nc) {      // 128 chunks x 64 codes
    __syncthreads();                      // prior chunk reads done
    #pragma unroll
    for (int i = 0; i < 16; ++i) {        // 4096 x 16B chunks / 256 thr
      int idx8 = tid + i * 256;
      int row = idx8 >> 6, c8 = (idx8 & 63) << 3;
      async_b128(cbb + (size_t)(nc * 64 + row) * 512 + c8, &Cs[row][c8]);
    }
    wait_async0();
    __syncthreads();
    #pragma unroll
    for (int t4 = 0; t4 < 4; ++t4) {      // 4 n-tiles of 16 codes
      int code = nc * 64 + t4 * 16 + llo;
      v8f acc = vzero();
      #pragma unroll
      for (int ks = 0; ks < 16; ++ks) {
        FragU bc;
        const bf16* p = &Cs[t4 * 16 + llo][ks * 32 + 8 * lhi];
        bc.h[0] = *(const v8bf*)p;
        bc.h[1] = *(const v8bf*)(p + 16);
        acc = wmma_bf16(az[ks].v, bc.v, acc);
      }
      float c2 = cn[code];
      #pragma unroll
      for (int r = 0; r < 8; ++r) {
        float d = c2 - 2.f * acc[r];
        if (d < mv[r]) { mv[r] = d; mi[r] = code; }
      }
    }
  }
  #pragma unroll
  for (int r = 0; r < 8; ++r) {
    #pragma unroll
    for (int off = 8; off >= 1; off >>= 1) {
      float ov = __shfl_xor(mv[r], off, 32);
      int   oi = __shfl_xor(mi[r], off, 32);
      if (ov < mv[r] || (ov == mv[r] && oi < mi[r])) { mv[r] = ov; mi[r] = oi; }
    }
    if (llo == 0) idx[tbase + r + 8 * lhi] = mi[r];
  }
}

// gather zq (transposed to BEFHW, exact f32 codebook rows), loss, idx as float
__global__ __launch_bounds__(256)
void vq_finalize_kernel(const float* __restrict__ z, const float* __restrict__ cb,
                        const int* __restrict__ idx, float* __restrict__ out_zq,
                        float* __restrict__ out_idx, float* __restrict__ lacc) {
  __shared__ float red8[8];
  int i = blockIdx.x * 256 + threadIdx.x;
  int t = i >> 9, e = i & 511;
  int b = t >> 12, f = (t >> 8) & 15, hw = t & 255;
  int id = idx[t];
  float c = cb[(size_t)id * 512 + e];
  float df = c - z[i];
  out_zq[(((size_t)b * 512 + e) * 16 + f) * 256 + hw] = c;
  if (e == 0) out_idx[t] = (float)id;
  float s = block_sum(df * df, red8, threadIdx.x);
  if (threadIdx.x == 0) atomicAdd(lacc, s);
}

__global__ void transpose_out_kernel(const float* __restrict__ tok, float* __restrict__ out) {
  int i = blockIdx.x * 256 + threadIdx.x;
  int t = i >> 9, e = i & 511;
  int b = t >> 12, f = (t >> 8) & 15, hw = t & 255;
  out[(((size_t)b * 512 + e) * 16 + f) * 256 + hw] = tok[i];
}

__global__ void loss_final_kernel(const float* __restrict__ lacc, float* __restrict__ out) {
  out[0] = lacc[0] * 1.25f / (8192.f * 512.f);
}

// ---------------------------------------------------------------------------
extern "C" void kernel_launch(void* const* d_in, const int* in_sizes, int n_in,
                              void* d_out, int out_size, void* d_ws, size_t ws_size,
                              hipStream_t stream) {
  (void)in_sizes; (void)n_in; (void)out_size; (void)ws_size;
  const float* x     = (const float*)d_in[0];
  const int*   ts    = (const int*)  d_in[1];
  const float* in_w  = (const float*)d_in[2];
  const float* in_b  = (const float*)d_in[3];
  const float* tw1   = (const float*)d_in[4];
  const float* tb1   = (const float*)d_in[5];
  const float* tw2   = (const float*)d_in[6];
  const float* tb2   = (const float*)d_in[7];
  const float* pos   = (const float*)d_in[8];
  const float* n1s   = (const float*)d_in[9];
  const float* n1b   = (const float*)d_in[10];
  const float* qkv_w = (const float*)d_in[11];
  const float* qkv_b = (const float*)d_in[12];
  const float* ow    = (const float*)d_in[13];
  const float* ob    = (const float*)d_in[14];
  const float* n2s   = (const float*)d_in[15];
  const float* n2b   = (const float*)d_in[16];
  const float* m1w   = (const float*)d_in[17];
  const float* m1b   = (const float*)d_in[18];
  const float* m2w   = (const float*)d_in[19];
  const float* m2b   = (const float*)d_in[20];
  const float* gns   = (const float*)d_in[21];
  const float* gnb   = (const float*)d_in[22];
  const float* opw   = (const float*)d_in[23];
  const float* opb   = (const float*)d_in[24];
  const float* qw    = (const float*)d_in[25];
  const float* qb    = (const float*)d_in[26];
  const float* cb    = (const float*)d_in[27];

  char* w = (char*)d_ws;
  float* hf    = (float*)(w);                               // 32 MB  f32 residual
  bf16*  xnb   = (bf16*) (w + ((size_t)32 << 20));          // 16 MB  LN/GN out
  bf16*  qkvb  = (bf16*) (w + ((size_t)64 << 20));          // 48 MB  qkv bf16
  bf16*  mlpt  = (bf16*) (w + ((size_t)120 << 20));         // 64 MB  MLP hidden
  bf16*  aob   = (bf16*) (w + ((size_t)160 << 20));         // 16 MB  attn out
  float* ptok  = (float*)(w + ((size_t)192 << 20));         // 16 MB  pred tokens
  float* ztok  = (float*)(w + ((size_t)208 << 20));         // 16 MB  z tokens
  char*  sm    = w + ((size_t)224 << 20);
  float* cn    = (float*)(sm);
  int*   idx   = (int*)  (sm + 32768);
  float* teraw = (float*)(sm + 65536);
  float* te1   = (float*)(sm + 73728);
  float* te    = (float*)(sm + 106496);
  float* lacc  = (float*)(sm + 114688);
  bf16*  wbf   = (bf16*) (w + ((size_t)226 << 20));         // 24 MB per-layer bf16 weights
  bf16*  cbbf  = (bf16*) (w + ((size_t)250 << 20));         // 8 MB  bf16 codebook
  bf16*  opwb  = (bf16*) (w + ((size_t)258 << 20));         // 1 MB
  bf16*  qwb   = (bf16*) (w + ((size_t)260 << 20));         // 0.5 MB
  // wbf sublayout (elements)
  bf16* wq  = wbf;               // 3145728
  bf16* wo  = wbf + 3145728;     // 1048576
  bf16* wm1 = wbf + 4194304;     // 4194304
  bf16* wm2 = wbf + 8388608;     // 4194304

  float* out_zq   = (float*)d_out;              // [2,512,16,16,16]
  float* out_pred = out_zq + 4194304;           // [2,512,16,16,16]
  float* out_loss = out_zq + 8388608;           // scalar
  float* out_idx  = out_zq + 8388609;           // [2,16,16,16] as float

  te_raw_kernel<<<4, 256, 0, stream>>>(ts, teraw, lacc);
  ts_mlp1_kernel<<<32, 256, 0, stream>>>(teraw, tw1, tb1, te1);
  ts_mlp2_kernel<<<8, 256, 0, stream>>>(te1, tw2, tb2, te);
  tokens_kernel<<<32768, 256, 0, stream>>>(x, in_w, in_b, te, pos, hf);
  cvt_bf16_kernel<<<2048, 256, 0, stream>>>(cb, cbbf, 8192 * 512);
  cvt_bf16_kernel<<<1024, 256, 0, stream>>>(opw, opwb, 512 * 1024);
  cvt_bf16_kernel<<<512, 256, 0, stream>>>(qw, qwb, 512 * 512);

  for (int l = 0; l < 12; ++l) {
    // per-layer weights -> bf16 (single f32 read, then bf16-resident in L2)
    cvt_bf16_kernel<<<2048, 256, 0, stream>>>(qkv_w + (size_t)l * 3145728, wq, 3145728);
    cvt_bf16_kernel<<<1024, 256, 0, stream>>>(ow + (size_t)l * 1048576, wo, 1048576);
    cvt_bf16_kernel<<<2048, 256, 0, stream>>>(m1w + (size_t)l * 4194304, wm1, 4194304);
    cvt_bf16_kernel<<<2048, 256, 0, stream>>>(m2w + (size_t)l * 4194304, wm2, 4194304);

    ln_kernel<<<8192, 256, 0, stream>>>(hf, n1s + l * 1024, n1b + l * 1024, xnb);
    gemm_bf16<0, false, true, true><<<dim3(24, 64), 256, 0, stream>>>(
        xnb, wq, qkv_b + l * 3072, nullptr, qkvb, 8192, 3072, 1024);
    attn_kernel<<<dim3(32, 16), 256, 0, stream>>>(qkvb, aob);
    gemm_bf16<0, true, true, false><<<dim3(8, 64), 256, 0, stream>>>(
        aob, wo, ob + l * 1024, hf, hf, 8192, 1024, 1024);
    ln_kernel<<<8192, 256, 0, stream>>>(hf, n2s + l * 1024, n2b + l * 1024, xnb);
    gemm_bf16<1, false, true, true><<<dim3(32, 64), 256, 0, stream>>>(
        xnb, wm1, m1b + l * 4096, nullptr, mlpt, 8192, 4096, 1024);
    gemm_bf16<0, true, true, false><<<dim3(8, 64), 256, 0, stream>>>(
        mlpt, wm2, m2b + l * 1024, hf, hf, 8192, 1024, 4096);
  }

  gn_silu_kernel<<<64, 256, 0, stream>>>(hf, gns, gnb, xnb);
  gemm_bf16<0, false, true, false><<<dim3(4, 64), 256, 0, stream>>>(
      xnb, opwb, opb, nullptr, ptok, 8192, 512, 1024);
  gemm_bf16<0, false, false, false><<<dim3(4, 64), 256, 0, stream>>>(
      ptok, qwb, qb, nullptr, ztok, 8192, 512, 512);
  transpose_out_kernel<<<16384, 256, 0, stream>>>(ptok, out_pred);
  cnorm_kernel<<<32, 256, 0, stream>>>(cb, cn);
  vq_argmin_kernel<<<64, 256, 0, stream>>>(ztok, cbbf, cn, idx);
  vq_finalize_kernel<<<16384, 256, 0, stream>>>(ztok, cb, idx, out_zq, out_idx, lacc);
  loss_final_kernel<<<1, 1, 0, stream>>>(lacc, out_loss);
}
